// MultiHeadAttention_80925773791289
// MI455X (gfx1250) — matrix operand order repaired and checked by reading, not compiled, not run on hardware
//
#include <hip/hip_runtime.h>

#define BATCH  4
#define SEQ    2048
#define DMODEL 1024
#define NH     16
#define DHEAD  64

typedef __attribute__((ext_vector_type(16))) __bf16 v16bf;
typedef __attribute__((ext_vector_type(8)))  float  v8f;
typedef __attribute__((ext_vector_type(4)))  unsigned int v4u;
typedef __attribute__((ext_vector_type(8)))  unsigned int v8u;

union Frag {
  v16bf v;
  unsigned int d[8];
};

__device__ __forceinline__ unsigned short f2bf(float f) {
  unsigned int u = __builtin_bit_cast(unsigned int, f);
  u += 0x7FFFu + ((u >> 16) & 1u);   // round-to-nearest-even
  return (unsigned short)(u >> 16);
}
__device__ __forceinline__ unsigned int pack2(unsigned short lo, unsigned short hi) {
  return (unsigned int)lo | ((unsigned int)hi << 16);
}
__device__ __forceinline__ v8f wmma_bf16(const Frag& a, const Frag& b, v8f c) {
  return __builtin_amdgcn_wmma_f32_16x16x32_bf16(false, a.v, false, b.v,
                                                 (short)0, c, false, false);
}
// CDNA5 async global->LDS copy (ASYNCcnt-tracked), 16B per lane.
__device__ __forceinline__ void async_ld_b128(unsigned lds_off, unsigned long long gaddr) {
  asm volatile("global_load_async_to_lds_b128 %0, %1, off"
               :: "v"(lds_off), "v"(gaddr) : "memory");
}
// CDNA5 Tensor Data Mover: DMA one 2D bf16 tile (rows x rowlen, row stride
// `stride` elements) from global to LDS. D# per CDNA5 ISA.
// Wave-level op (EXEC ignored) tracked on TENSORcnt — issue from ONE wave.
__device__ __forceinline__ void tdm_load_tile_bf16(unsigned lds_addr, unsigned long long ga,
                                                   unsigned rowlen, unsigned rows,
                                                   unsigned stride) {
  v4u g0;
  g0[0] = 1u;                                    // count=1, user descriptor
  g0[1] = lds_addr;                              // lds_addr (bytes)
  g0[2] = (unsigned)ga;                          // global_addr[31:0]
  g0[3] = (unsigned)(ga >> 32) | (2u << 30);     // global_addr[56:32] | type=2
  v8u g1;
  g1[0] = 1u << 16;                              // data_size=2B; wg_mask=0
  g1[1] = rowlen << 16;                          // tensor_dim0[15:0] @bits63:48
  g1[2] = (unsigned)SEQ << 16;                   // tensor_dim1[15:0] @bits95:80
  g1[3] = rowlen << 16;                          // tile_dim0 @bits127:112
  g1[4] = rows;                                  // tile_dim1 @bits143:128
  g1[5] = stride;                                // tensor_dim0_stride[31:0]
  g1[6] = 0u;
  g1[7] = 0u;
  asm volatile("tensor_load_to_lds %0, %1" :: "s"(g0), "s"(g1) : "memory");
}

// ---------------------------------------------------------------------------
// Generic 128x64-tile GEMM, bf16 WMMA, f32 accumulate.
//   A: [M x K] row-major, TA = float (convert on load) or ushort (bf16 raw)
//   B: [K x N] row-major f32, converted to bf16 and transposed into LDS
//   MODE 0: out bf16 scattered to [B,H,S,64]      (Q / K projections)
//   MODE 1: out bf16 scattered to [B,H,64,S]      (V projection, transposed)
//   MODE 2: out f32 row-major [M,N]               (final output projection)
// ---------------------------------------------------------------------------
template <typename TA, int MODE>
__global__ __launch_bounds__(256) void gemm_wmma_kernel(
    const TA* __restrict__ A, const float* __restrict__ Bm,
    void* __restrict__ Out, int Kdim, int Ndim, float scale) {
  __shared__ unsigned int Asd[128 * 16];   // 128 rows x 32 bf16 (dword pairs)
  __shared__ unsigned int Bsd[64 * 16];    // transposed: [n][k], 32 bf16 per n
  unsigned short* Bss = (unsigned short*)Bsd;

  const int tid  = threadIdx.x;
  const int lane = tid & 31;
  const int wv   = tid >> 5;
  const int wr   = wv >> 1;                // 0..3
  const int wc   = wv & 1;                 // 0..1
  const int mBase = wr * 32;
  const int nBase = wc * 32;
  const int kh = lane >> 4;
  const int lr = lane & 15;
  const long m0 = (long)blockIdx.y * 128;
  const long n0 = (long)blockIdx.x * 64;

  v8f acc[2][2];
  for (int a = 0; a < 2; ++a)
    for (int b = 0; b < 2; ++b)
      for (int i = 0; i < 8; ++i) acc[a][b][i] = 0.0f;

  for (int kk = 0; kk < Kdim; kk += 32) {
    __syncthreads();
    {  // A tile: 128 x 32, each thread loads 16 elements
      const int row  = tid >> 1;
      const int half = (tid & 1) * 16;
      if constexpr (sizeof(TA) == 4) {
        const float* src = (const float*)A + (m0 + row) * (long)Kdim + kk + half;
#pragma unroll
        for (int t = 0; t < 8; ++t)
          Asd[row * 16 + (half >> 1) + t] = pack2(f2bf(src[2 * t]), f2bf(src[2 * t + 1]));
      } else {
        const unsigned int* src =
            (const unsigned int*)((const unsigned short*)A + (m0 + row) * (long)Kdim + kk + half);
#pragma unroll
        for (int t = 0; t < 8; ++t)
          Asd[row * 16 + (half >> 1) + t] = src[t];
      }
    }
    {  // B tile: 32 x 64 f32, store transposed [n][k] as bf16
      const int krow = tid >> 3;           // 0..31
      const int nb   = (tid & 7) * 8;      // 0,8,..,56
      const float* src = Bm + (long)(kk + krow) * Ndim + n0 + nb;
#pragma unroll
      for (int j = 0; j < 8; ++j)
        Bss[(nb + j) * 32 + krow] = f2bf(src[j]);
    }
    __syncthreads();

    Frag fa[2], fb[2];
#pragma unroll
    for (int ti = 0; ti < 2; ++ti) {
      const int row = mBase + ti * 16 + lr;
#pragma unroll
      for (int j = 0; j < 8; ++j) {
        // A 16x32 bf16 layout: K = (j<4?0:16) + 8*(lane>>4) + 2*(j&3)
        const int kbd = ((j < 4) ? 0 : 8) + 4 * kh + (j & 3);   // dword index
        fa[ti].d[j] = Asd[row * 16 + kbd];
      }
    }
#pragma unroll
    for (int tj = 0; tj < 2; ++tj) {
      const int n = nBase + tj * 16 + lr;
#pragma unroll
      for (int j = 0; j < 8; ++j)
        // B 32x16 bf16 layout: K = (j<4?0:16) + 8*(lane>>4) + 2*(j&3)
        fb[tj].d[j] = Bsd[n * 16 + ((j < 4) ? 0 : 8) + 4 * kh + (j & 3)];
    }
#pragma unroll
    for (int ti = 0; ti < 2; ++ti)
#pragma unroll
      for (int tj = 0; tj < 2; ++tj)
        acc[ti][tj] = wmma_bf16(fa[ti], fb[tj], acc[ti][tj]);
  }

#pragma unroll
  for (int ti = 0; ti < 2; ++ti)
#pragma unroll
    for (int tj = 0; tj < 2; ++tj)
#pragma unroll
      for (int i = 0; i < 8; ++i) {
        const long gm = m0 + mBase + ti * 16 + i + 8 * kh;   // row (C layout)
        const long gn = n0 + nBase + tj * 16 + lr;           // col
        const float val = acc[ti][tj][i] * scale;
        if constexpr (MODE == 0) {
          const long b = gm / SEQ, s = gm % SEQ, h = gn >> 6, dk = gn & 63;
          ((unsigned short*)Out)[((b * NH + h) * SEQ + s) * 64 + dk] = f2bf(val);
        } else if constexpr (MODE == 1) {
          const long b = gm / SEQ, s = gm % SEQ, h = gn >> 6, dv = gn & 63;
          ((unsigned short*)Out)[((b * NH + h) * 64 + dv) * SEQ + s] = f2bf(val);
        } else {
          ((float*)Out)[gm * Ndim + gn] = val;
        }
      }
}

// ---------------------------------------------------------------------------
// Causal flash attention, block-cooperative K/V staging.
//   Block = 128 threads (4 waves) on one (b,h); wave w owns q rows
//   [q0_block + 16w, +16). K-tile (32x64) staged by the Tensor Data Mover
//   (TENSORcnt), V-tile (64x32, V^T layout) by async b128 loads (ASYNCcnt);
//   both double-buffered and overlapped with the WMMA pipeline.
//   Q,K: bf16 [B,H,S,64]; V: bf16 [B,H,64,S]; out bf16 [B*S, H*64].
//   1/sqrt(dk) folded into Q.
// ---------------------------------------------------------------------------
__global__ __launch_bounds__(128) void flash_attn_kernel(
    const unsigned short* __restrict__ Qbf, const unsigned short* __restrict__ Kbf,
    const unsigned short* __restrict__ Vtb, unsigned short* __restrict__ Obf) {
  __shared__ unsigned short Ktile[2][32 * 64];   // [pos][dk], 4KB per buffer
  __shared__ unsigned short Vtile[2][64 * 32];   // [dv][pos], 4KB per buffer
  __shared__ unsigned int   Psd[4][256];         // per-wave 16x32 bf16 P staging

  const int tid  = threadIdx.x;
  const int lane = tid & 31;
  const int wave = tid >> 5;
  const int kh   = lane >> 4;
  const int lr   = lane & 15;
  const int bh   = blockIdx.y;
  const int q0b  = blockIdx.x * 64;       // block's first query row
  const int q0   = q0b + wave * 16;       // this wave's query tile

  const unsigned short* Kg = Kbf + (long)bh * SEQ * 64;
  const unsigned short* Vg = Vtb + (long)bh * 64 * SEQ;
  const unsigned int*   Qd = (const unsigned int*)Qbf + (long)bh * SEQ * 32;

  const unsigned kbase[2] = {(unsigned)(uintptr_t)&Ktile[0][0],
                             (unsigned)(uintptr_t)&Ktile[1][0]};
  const unsigned vbase[2] = {(unsigned)(uintptr_t)&Vtile[0][0],
                             (unsigned)(uintptr_t)&Vtile[1][0]};

  // Issue staging of one 32-key block into LDS buffer `b`:
  //   K tile: one TDM descriptor, issued by wave 0 only (TDM ignores EXEC).
  //   V tile: per-lane async b128 copies (2 instructions -> ASYNCcnt += 2).
  auto issue_tiles = [&](int j0, int b) {
    if (wave == 0)
      tdm_load_tile_bf16(kbase[b],
                         (unsigned long long)(uintptr_t)(Kg + (long)j0 * 64),
                         /*rowlen=*/64, /*rows=*/32, /*stride=*/64);
    for (int c = tid; c < 256; c += 128) {        // V^T: 64 rows x 64B = 256 chunks
      const int dv = c >> 2, part = c & 3;
      async_ld_b128(vbase[b] + c * 16,
                    (unsigned long long)(uintptr_t)(Vg + (long)dv * SEQ + j0 + part * 8));
    }
  };

  Frag a0, a1;                             // Q tile 16x64 -> two A fragments
  {
    const long qrow = q0 + lr;
#pragma unroll
    for (int j = 0; j < 8; ++j) {
      const int kbd = ((j < 4) ? 0 : 8) + 4 * kh + (j & 3);
      a0.d[j] = Qd[qrow * 32 + kbd];
      a1.d[j] = Qd[qrow * 32 + 16 + kbd];
    }
  }

  float mrow[8], lrow[8];
  v8f o[4];
#pragma unroll
  for (int i = 0; i < 8; ++i) { mrow[i] = -1e30f; lrow[i] = 0.0f; }
#pragma unroll
  for (int t = 0; t < 4; ++t)
#pragma unroll
    for (int i = 0; i < 8; ++i) o[t][i] = 0.0f;

  unsigned short* psh = (unsigned short*)(&Psd[wave][0]);

  const int nblocks = q0b / 32 + 2;        // covers keys up to q0b+63
  int buf = 0;
  issue_tiles(0, 0);

  for (int jb = 0; jb < nblocks; ++jb) {
    const int j0 = jb * 32;
    const bool have_next = (jb + 1 < nblocks);
    if (have_next) issue_tiles(j0 + 32, buf ^ 1);
    if (wave == 0) {                       // TDM completion for current K tile
      if (have_next) __builtin_amdgcn_s_wait_tensorcnt(1);
      else           __builtin_amdgcn_s_wait_tensorcnt(0);
    }
    if (have_next) asm volatile("s_wait_asynccnt 0x2" ::: "memory");
    else           asm volatile("s_wait_asynccnt 0x0" ::: "memory");
    __syncthreads();                       // current buffers visible to all waves

    if (j0 <= q0 + 15) {                   // wave-uniform causal skip
      const unsigned int* Kt = (const unsigned int*)&Ktile[buf][0];
      const unsigned int* Vt = (const unsigned int*)&Vtile[buf][0];

      Frag b0a, b0b, b1a, b1b;
#pragma unroll
      for (int j = 0; j < 8; ++j) {
        b0a.d[j] = Kt[lr * 32 + ((j < 4) ? 0 : 8) + 4 * kh + (j & 3)];
        b0b.d[j] = Kt[lr * 32 + 16 + ((j < 4) ? 0 : 8) + 4 * kh + (j & 3)];
        b1a.d[j] = Kt[(lr + 16) * 32 + ((j < 4) ? 0 : 8) + 4 * kh + (j & 3)];
        b1b.d[j] = Kt[(lr + 16) * 32 + 16 + ((j < 4) ? 0 : 8) + 4 * kh + (j & 3)];
      }
      v8f s0, s1;
#pragma unroll
      for (int i = 0; i < 8; ++i) { s0[i] = 0.0f; s1[i] = 0.0f; }
      s0 = wmma_bf16(a0, b0a, s0);
      s0 = wmma_bf16(a1, b0b, s0);
      s1 = wmma_bf16(a0, b1a, s1);
      s1 = wmma_bf16(a1, b1b, s1);

      // causal mask + online softmax (row spread over 16 lanes in C layout)
#pragma unroll
      for (int i = 0; i < 8; ++i) {
        const int row = q0 + i + 8 * kh;
        float v0 = (j0 + lr > row) ? -1e30f : s0[i];
        float v1 = (j0 + 16 + lr > row) ? -1e30f : s1[i];
        float ml = fmaxf(v0, v1);
#pragma unroll
        for (int off = 8; off >= 1; off >>= 1)
          ml = fmaxf(ml, __shfl_xor(ml, off, 32));
        const float mnew = fmaxf(mrow[i], ml);
        const float corr = __expf(mrow[i] - mnew);
        const float p0 = __expf(v0 - mnew);
        const float p1 = __expf(v1 - mnew);
        float ps = p0 + p1;
#pragma unroll
        for (int off = 8; off >= 1; off >>= 1)
          ps += __shfl_xor(ps, off, 32);
        lrow[i] = lrow[i] * corr + ps;
        mrow[i] = mnew;
#pragma unroll
        for (int t = 0; t < 4; ++t) o[t][i] *= corr;
        psh[(i + 8 * kh) * 32 + lr]      = f2bf(p0);
        psh[(i + 8 * kh) * 32 + lr + 16] = f2bf(p1);
      }
      asm volatile("s_wait_dscnt 0x0" ::: "memory");  // P store -> P read, same wave

      Frag p;                                         // P as A fragment 16x32
#pragma unroll
      for (int j = 0; j < 8; ++j) {
        const int kbd = ((j < 4) ? 0 : 8) + 4 * kh + (j & 3);
        p.d[j] = Psd[wave][lr * 16 + kbd];
      }
#pragma unroll
      for (int t = 0; t < 4; ++t) {                   // O += P @ V
        Frag vb;
        const int dv = t * 16 + lr;
#pragma unroll
        for (int j = 0; j < 8; ++j)
          vb.d[j] = Vt[dv * 16 + ((j < 4) ? 0 : 8) + 4 * kh + (j & 3)];
        o[t] = wmma_bf16(p, vb, o[t]);
      }
    }

    __syncthreads();                       // buffers free before next overwrite
    buf ^= 1;
  }

  const int b = bh >> 4;
  const int h = bh & 15;
#pragma unroll
  for (int t = 0; t < 4; ++t)
#pragma unroll
    for (int i = 0; i < 8; ++i) {
      const int row = q0 + i + 8 * kh;
      const int col = h * 64 + t * 16 + lr;
      const float val = o[t][i] / lrow[i];
      Obf[((long)(b * SEQ + row)) * (NH * DHEAD) + col] = f2bf(val);
    }
}

// ---------------------------------------------------------------------------
extern "C" void kernel_launch(void* const* d_in, const int* in_sizes, int n_in,
                              void* d_out, int out_size, void* d_ws, size_t ws_size,
                              hipStream_t stream) {
  (void)in_sizes; (void)n_in; (void)out_size; (void)ws_size;
  const float* query = (const float*)d_in[0];
  const float* key_t = (const float*)d_in[1];
  const float* value = (const float*)d_in[2];
  // d_in[3] = mask: causal, implemented analytically in-kernel
  const float* w_q = (const float*)d_in[4];
  const float* w_k = (const float*)d_in[5];
  const float* w_v = (const float*)d_in[6];
  const float* w_o = (const float*)d_in[7];

  char* ws = (char*)d_ws;
  const size_t headBytes = (size_t)BATCH * NH * SEQ * DHEAD * sizeof(unsigned short); // 16 MB
  unsigned short* Qbf = (unsigned short*)(ws);
  unsigned short* Kbf = (unsigned short*)(ws + headBytes);
  unsigned short* Vtb = (unsigned short*)(ws + 2 * headBytes);
  unsigned short* Obf = (unsigned short*)(ws + 3 * headBytes);

  dim3 gg(DMODEL / 64, (BATCH * SEQ) / 128);
  // scale 1/sqrt(64) folded into Q projection
  gemm_wmma_kernel<float, 0><<<gg, 256, 0, stream>>>(query, w_q, Qbf, DMODEL, DMODEL, 0.125f);
  gemm_wmma_kernel<float, 0><<<gg, 256, 0, stream>>>(key_t, w_k, Kbf, DMODEL, DMODEL, 1.0f);
  gemm_wmma_kernel<float, 1><<<gg, 256, 0, stream>>>(value, w_v, Vtb, DMODEL, DMODEL, 1.0f);
  flash_attn_kernel<<<dim3(SEQ / 64, BATCH * NH), 128, 0, stream>>>(Qbf, Kbf, Vtb, Obf);
  gemm_wmma_kernel<unsigned short, 2><<<gg, 256, 0, stream>>>(Obf, w_o, d_out, DMODEL, DMODEL, 1.0f);
}
